// Block_85169201479865
// MI455X (gfx1250) — matrix operand order repaired and checked by reading, not compiled, not run on hardware
//
#include <hip/hip_runtime.h>
#include <hip/hip_bf16.h>
#include <math.h>

// ---------------- problem constants ----------------
constexpr int kB   = 2;
constexpr int kS   = 2048;
constexpr int kD   = 1024;
constexpr int kH   = 16;
constexpr int kHD  = 64;
constexpr int kFFN = 2048;
constexpr int kE   = 8;
constexpr int kT   = kB * kS;          // 4096 tokens
constexpr int kCAP = 1280;             // ceil(T*2*1.25/8)

typedef __attribute__((ext_vector_type(16))) __bf16 v16bf;
typedef __attribute__((ext_vector_type(8)))  float  v8f;

__device__ __forceinline__ v8f wmma_bf16(const v16bf& a, const v16bf& b, const v8f& c) {
  return __builtin_amdgcn_wmma_f32_16x16x32_bf16(false, a, false, b, (short)0, c, false, false);
}

// ---- CDNA5 async LDS-DMA path, guarded so either toolchain compiles ----
#if defined(__gfx1250__) && __has_builtin(__builtin_amdgcn_global_load_async_to_lds_b128)
  #define USE_ASYNC_COPY 1
#else
  #define USE_ASYNC_COPY 0
#endif

#if USE_ASYNC_COPY
typedef int v4i __attribute__((ext_vector_type(4)));
typedef __attribute__((address_space(1))) v4i* as1_v4i;   // global
typedef __attribute__((address_space(3))) v4i* as3_v4i;   // LDS
#endif

__device__ __forceinline__ void async_wait_all() {
#if USE_ASYNC_COPY
  #if __has_builtin(__builtin_amdgcn_s_wait_asynccnt)
    __builtin_amdgcn_s_wait_asynccnt(0);
  #else
    asm volatile("s_wait_asynccnt 0" ::: "memory");
  #endif
#endif
}

// ---------------- fp32 -> bf16 bulk convert (weights, once per call) ----------------
__global__ __launch_bounds__(256)
void convert_f32_to_bf16(const float* __restrict__ src, __bf16* __restrict__ dst, int n) {
  const int i = (blockIdx.x * 256 + threadIdx.x) * 8;
  if (i >= n) return;
  const float4 f0 = ((const float4*)(src + i))[0];
  const float4 f1 = ((const float4*)(src + i))[1];
  __bf16 t[8] = {(__bf16)f0.x, (__bf16)f0.y, (__bf16)f0.z, (__bf16)f0.w,
                 (__bf16)f1.x, (__bf16)f1.y, (__bf16)f1.z, (__bf16)f1.w};
  *(uint4*)(dst + i) = *(const uint4*)t;
}

// ---------------- RMSNorm (fp32 in, bf16 out) ----------------
__global__ __launch_bounds__(256)
void rmsnorm_to_bf16(const float* __restrict__ x, const float* __restrict__ g,
                     __bf16* __restrict__ out) {
  __shared__ float red[256];
  const int row = blockIdx.x;
  const float* xr = x + (size_t)row * kD;
  float s = 0.f;
  for (int c = threadIdx.x; c < kD; c += 256) { float v = xr[c]; s += v * v; }
  red[threadIdx.x] = s; __syncthreads();
  for (int st = 128; st > 0; st >>= 1) {
    if (threadIdx.x < st) red[threadIdx.x] += red[threadIdx.x + st];
    __syncthreads();
  }
  const float inv = rsqrtf(red[0] / (float)kD + 1e-6f);
  __bf16* o = out + (size_t)row * kD;
  for (int c = threadIdx.x; c < kD; c += 256) o[c] = (__bf16)(xr[c] * g[c] * inv);
}

// ---------------- generic WMMA GEMM, all-bf16 operands, double-buffered LDS ----------------
// A: bf16 [M,K] row-major (+bz*M*K), Bw: bf16 [K,N] row-major (+bz*K*N)
// MODE 0: C bf16 store          MODE 1: C fp32 = resid + acc
// MODE 2: C bf16 relu (+bz)     MODE 3: atomicAdd(out[tok[slot]] += acc*gate)
template<int MODE>
__global__ __launch_bounds__(256)
void gemm_bf16_wmma(const __bf16* __restrict__ A, const __bf16* __restrict__ Bw,
                    void* __restrict__ Cptr, const float* __restrict__ resid,
                    const float* __restrict__ gates, const int* __restrict__ toks,
                    int M, int N, int K) {
  __shared__ __attribute__((aligned(16))) __bf16 Als[2][128][40];
  __shared__ __attribute__((aligned(16))) __bf16 Bls[2][32][72];
  const int tid  = threadIdx.x;
  const int lane = tid & 31, wave = tid >> 5;
  const int half = lane >> 4, nlo = lane & 15;
  const int rowBase = blockIdx.y * 128;
  const int colBase = blockIdx.x * 64;
  const int bz = blockIdx.z;
  const __bf16* Ab = A  + (size_t)bz * M * K;
  const __bf16* Bb = Bw + (size_t)bz * K * N;

  // per-thread staging coordinates (A: 128x32 tile, B: 32x64 tile, 16B moves)
  const int ra = tid >> 1, c0a = (tid & 1) << 4;   // A: 2 x b128 per thread
  const int rb = tid >> 3, c0b = (tid & 7) << 3;   // B: 1 x b128 per thread
  const __bf16* Agp = Ab + (size_t)(rowBase + ra) * K + c0a;
  const __bf16* Bgp = Bb + (size_t)rb * N + colBase + c0b;

  auto stage = [&](int kt, int buf) {
    const __bf16* ga = Agp + kt * 32;
    const __bf16* gb = Bgp + (size_t)kt * 32 * N;
#if USE_ASYNC_COPY
    __builtin_amdgcn_global_load_async_to_lds_b128(
        (as1_v4i)ga, (as3_v4i)&Als[buf][ra][c0a], 0, 0);
    __builtin_amdgcn_global_load_async_to_lds_b128(
        (as1_v4i)ga, (as3_v4i)&Als[buf][ra][c0a], 16, 0);
    __builtin_amdgcn_global_load_async_to_lds_b128(
        (as1_v4i)gb, (as3_v4i)&Bls[buf][rb][c0b], 0, 0);
#else
    const uint4* sa = (const uint4*)ga;
    uint4* da = (uint4*)&Als[buf][ra][c0a];
    da[0] = sa[0]; da[1] = sa[1];
    *(uint4*)&Bls[buf][rb][c0b] = *(const uint4*)gb;
#endif
  };

  v8f acc[4];
  #pragma unroll
  for (int n = 0; n < 4; n++)
    #pragma unroll
    for (int j = 0; j < 8; j++) acc[n][j] = 0.f;

  const int NT = K >> 5;
  stage(0, 0);
  async_wait_all();
  __syncthreads();

  int buf = 0;
  for (int kt = 0; kt < NT; kt++) {
    if (kt + 1 < NT) stage(kt + 1, buf ^ 1);  // overlap next-stage DMA with WMMA

    // A fragment (16x32, contiguous k = half*16 + s per lane)
    v16bf af;
    const int am = wave * 16 + nlo;
    #pragma unroll
    for (int i = 0; i < 8; i++) {
      const int kk = half * 16 + 2 * i;
      af[2*i]   = Als[buf][am][kk];
      af[2*i+1] = Als[buf][am][kk + 1];
    }
    #pragma unroll
    for (int nb = 0; nb < 4; nb++) {
      v16bf bf;  // B 32x16, same contiguous k = half*16 + s as A
      #pragma unroll
      for (int i = 0; i < 8; i++) {
        const int kk = half * 16 + 2 * i;
        bf[2*i]   = Bls[buf][kk][nb * 16 + nlo];
        bf[2*i+1] = Bls[buf][kk + 1][nb * 16 + nlo];
      }
      acc[nb] = wmma_bf16(af, bf, acc[nb]);
    }
    async_wait_all();
    __syncthreads();
    buf ^= 1;
  }

  // epilogue: D layout -> vgpr j holds row half*8+j, col = lane&15
  #pragma unroll
  for (int j = 0; j < 8; j++) {
    const int lr = rowBase + wave * 16 + half * 8 + j;
    #pragma unroll
    for (int nb = 0; nb < 4; nb++) {
      const int gc = colBase + nb * 16 + nlo;
      const float v = acc[nb][j];
      if constexpr (MODE == 0) {
        ((__bf16*)Cptr)[(size_t)lr * N + gc] = (__bf16)v;
      } else if constexpr (MODE == 1) {
        const size_t idx = (size_t)lr * N + gc;
        ((float*)Cptr)[idx] = resid[idx] + v;
      } else if constexpr (MODE == 2) {
        ((__bf16*)Cptr)[(size_t)bz * M * N + (size_t)lr * N + gc] = (__bf16)fmaxf(v, 0.f);
      } else {
        const int slot = bz * M + lr;
        const float gt = gates[slot];
        const int  tok = toks[slot];
        atomicAdd(&((float*)Cptr)[(size_t)tok * N + gc], v * gt);
      }
    }
  }
}

// ---------------- RoPE + [T,3D] -> [b,h,s,hd] repack ----------------
__global__ __launch_bounds__(256)
void rope_repack(const __bf16* __restrict__ qkv, __bf16* __restrict__ qh,
                 __bf16* __restrict__ kh, __bf16* __restrict__ vh) {
  const int gid = blockIdx.x * blockDim.x + threadIdx.x;
  if (gid >= kT * kH * (kHD / 2)) return;
  const int pair = gid & 31;
  const int h = (gid >> 5) & (kH - 1);
  const int t = gid >> 9;                 // /(32*16)
  const int s = t & (kS - 1);
  const int b = t >> 11;                  // /kS
  const size_t src = (size_t)t * 3 * kD + h * kHD + 2 * pair;
  const float qr = (float)qkv[src],          qi = (float)qkv[src + 1];
  const float kr = (float)qkv[src + kD],     ki = (float)qkv[src + kD + 1];
  const float vr = (float)qkv[src + 2 * kD], vi = (float)qkv[src + 2 * kD + 1];
  const float th = __expf(-(float)(2 * pair) * (9.210340371976184f / 64.f)); // 10000^(-2i/64)
  float sn, cs; __sincosf((float)s * th, &sn, &cs);
  const size_t dst = (((size_t)(b * kH + h)) * kS + s) * kHD + 2 * pair;
  qh[dst]   = (__bf16)(qr * cs - qi * sn);
  qh[dst+1] = (__bf16)(qr * sn + qi * cs);
  kh[dst]   = (__bf16)(kr * cs - ki * sn);
  kh[dst+1] = (__bf16)(kr * sn + ki * cs);
  vh[dst]   = (__bf16)vr;
  vh[dst+1] = (__bf16)vi;
}

// ---------------- flash-style causal attention (1 wave / 16 q rows) ----------------
constexpr int KC = 512;  // key chunk
__global__ __launch_bounds__(32)
void attn_wmma(const __bf16* __restrict__ qh, const __bf16* __restrict__ kh,
               const __bf16* __restrict__ vh, __bf16* __restrict__ y) {
  __shared__ float sc[16 * KC];
  __shared__ float rowM[16], rowS[16], rowF[16];
  const int lane = threadIdx.x;
  const int half = lane >> 4, nlo = lane & 15;
  const int qb = blockIdx.x & (kS / 16 - 1);
  const int h  = (blockIdx.x / (kS / 16)) & (kH - 1);
  const int b  = blockIdx.x / (kS / 16 * kH);
  const int qbase = qb * 16;
  const size_t headOff = ((size_t)(b * kH + h)) * kS * kHD;
  const __bf16* Q  = qh + headOff;
  const __bf16* Kp = kh + headOff;
  const __bf16* V  = vh + headOff;

  if (lane < 16) { rowM[lane] = -INFINITY; rowS[lane] = 0.f; }

  v16bf qf[2];
  #pragma unroll
  for (int s2 = 0; s2 < 2; s2++)
    #pragma unroll
    for (int i = 0; i < 8; i++) {
      const int kk = half * 16 + 2 * i;
      qf[s2][2*i]   = Q[(size_t)(qbase + nlo) * kHD + s2 * 32 + kk];
      qf[s2][2*i+1] = Q[(size_t)(qbase + nlo) * kHD + s2 * 32 + kk + 1];
    }

  v8f acc4[4];
  #pragma unroll
  for (int n = 0; n < 4; n++)
    #pragma unroll
    for (int j = 0; j < 8; j++) acc4[n][j] = 0.f;

  const int kmax = qbase + 16;  // exclusive causal key bound for this block
  for (int c0 = 0; c0 < kmax; c0 += KC) {
    const int kbmax = (min(KC, kmax - c0) + 31) >> 5;
    // ---- scores: S = (Q K^T) / sqrt(hd) ----
    for (int kb = 0; kb < kbmax; kb++) {
      #pragma unroll
      for (int nt = 0; nt < 2; nt++) {
        v8f acc;
        #pragma unroll
        for (int j = 0; j < 8; j++) acc[j] = 0.f;
        const int key = c0 + kb * 32 + nt * 16 + nlo;
        #pragma unroll
        for (int s2 = 0; s2 < 2; s2++) {
          v16bf bf;
          #pragma unroll
          for (int i = 0; i < 8; i++) {
            const int kk = half * 16 + 2 * i;
            bf[2*i]   = Kp[(size_t)key * kHD + s2 * 32 + kk];
            bf[2*i+1] = Kp[(size_t)key * kHD + s2 * 32 + kk + 1];
          }
          acc = wmma_bf16(qf[s2], bf, acc);
        }
        const int col = kb * 32 + nt * 16 + nlo;
        #pragma unroll
        for (int j = 0; j < 8; j++) sc[(half * 8 + j) * KC + col] = acc[j] * 0.125f;
      }
    }
    // ---- online softmax over this chunk ----
    const int ncc = kbmax * 32;
    for (int r = 0; r < 16; r++) {
      const int qidx = qbase + r;
      float cm = -INFINITY;
      for (int c = lane; c < ncc; c += 32)
        if (c0 + c <= qidx) cm = fmaxf(cm, sc[r * KC + c]);
      #pragma unroll
      for (int st = 16; st > 0; st >>= 1) cm = fmaxf(cm, __shfl_xor(cm, st, 32));
      const float oldM = rowM[r];
      const float nm = fmaxf(oldM, cm);
      const float f  = __expf(oldM - nm);     // exp(-inf)=0 on first chunk
      float sum = 0.f;
      for (int c = lane; c < ncc; c += 32) {
        const float e = (c0 + c <= qidx) ? __expf(sc[r * KC + c] - nm) : 0.f;
        sc[r * KC + c] = e;
        sum += e;
      }
      #pragma unroll
      for (int st = 16; st > 0; st >>= 1) sum += __shfl_xor(sum, st, 32);
      if (lane == 0) { rowS[r] = rowS[r] * f + sum; rowM[r] = nm; rowF[r] = f; }
    }
    #pragma unroll
    for (int nt = 0; nt < 4; nt++)
      #pragma unroll
      for (int j = 0; j < 8; j++) acc4[nt][j] *= rowF[half * 8 + j];
    // ---- P @ V accumulate ----
    for (int kb = 0; kb < kbmax; kb++) {
      v16bf af;
      #pragma unroll
      for (int i = 0; i < 8; i++) {
        const int kk = half * 16 + 2 * i;
        af[2*i]   = (__bf16)sc[nlo * KC + kb * 32 + kk];
        af[2*i+1] = (__bf16)sc[nlo * KC + kb * 32 + kk + 1];
      }
      #pragma unroll
      for (int nt = 0; nt < 4; nt++) {
        v16bf bf;
        #pragma unroll
        for (int i = 0; i < 8; i++) {
          const int kk = half * 16 + 2 * i;
          bf[2*i]   = V[(size_t)(c0 + kb * 32 + kk) * kHD + nt * 16 + nlo];
          bf[2*i+1] = V[(size_t)(c0 + kb * 32 + kk + 1) * kHD + nt * 16 + nlo];
        }
        acc4[nt] = wmma_bf16(af, bf, acc4[nt]);
      }
    }
  }
  #pragma unroll
  for (int j = 0; j < 8; j++) {
    const float rinv = 1.f / rowS[half * 8 + j];
    const int q = qbase + half * 8 + j;
    const size_t t = (size_t)b * kS + q;
    #pragma unroll
    for (int nt = 0; nt < 4; nt++)
      y[t * kD + h * kHD + nt * 16 + nlo] = (__bf16)(acc4[nt][j] * rinv);
  }
}

// ---------------- router: softmax + top-2 + capacity assign ----------------
__global__ void zero_cnt(int* cnt) { if (threadIdx.x < kE) cnt[threadIdx.x] = 0; }

__global__ __launch_bounds__(256)
void router_topk(const __bf16* __restrict__ x2, const float* __restrict__ wr,
                 int* __restrict__ cnt, int* __restrict__ toks, float* __restrict__ gates) {
  const int t = blockIdx.x * blockDim.x + threadIdx.x;
  if (t >= kT) return;
  float lg[kE];
  #pragma unroll
  for (int e = 0; e < kE; e++) lg[e] = 0.f;
  const __bf16* xr = x2 + (size_t)t * kD;
  for (int d = 0; d < kD; d++) {
    const float xv = (float)xr[d];
    const float* w = wr + (size_t)d * kE;
    #pragma unroll
    for (int e = 0; e < kE; e++) lg[e] += xv * w[e];
  }
  float mx = lg[0];
  #pragma unroll
  for (int e = 1; e < kE; e++) mx = fmaxf(mx, lg[e]);
  float p[kE]; float sum = 0.f;
  #pragma unroll
  for (int e = 0; e < kE; e++) { p[e] = __expf(lg[e] - mx); sum += p[e]; }
  #pragma unroll
  for (int e = 0; e < kE; e++) p[e] /= sum;
  int e0 = 0;
  #pragma unroll
  for (int e = 1; e < kE; e++) if (p[e] > p[e0]) e0 = e;
  int e1 = (e0 == 0) ? 1 : 0;
  #pragma unroll
  for (int e = 0; e < kE; e++) if (e != e0 && p[e] > p[e1]) e1 = e;
  int pos = atomicAdd(&cnt[e0], 1);
  if (pos < kCAP) { toks[e0 * kCAP + pos] = t; gates[e0 * kCAP + pos] = p[e0]; }
  pos = atomicAdd(&cnt[e1], 1);
  if (pos < kCAP) { toks[e1 * kCAP + pos] = t; gates[e1 * kCAP + pos] = p[e1]; }
}

// ---------------- gather expert inputs (zero-fill unused slots) ----------------
__global__ __launch_bounds__(256)
void gather_xe(const __bf16* __restrict__ x2, const int* __restrict__ cnt,
               int* __restrict__ toks, float* __restrict__ gates, __bf16* __restrict__ xe) {
  const int slot = blockIdx.x;                // e*CAP + c
  const int e = slot / kCAP, c = slot - e * kCAP;
  const bool valid = c < min(cnt[e], kCAP);
  if (!valid && threadIdx.x == 0) { toks[slot] = 0; gates[slot] = 0.f; }
  const int t = valid ? toks[slot] : 0;
  const __bf16* src = x2 + (size_t)t * kD;
  __bf16* dst = xe + (size_t)slot * kD;
  for (int d = threadIdx.x; d < kD; d += 256)
    dst[d] = valid ? src[d] : (__bf16)0.f;
}

// ---------------- launcher ----------------
extern "C" void kernel_launch(void* const* d_in, const int* in_sizes, int n_in,
                              void* d_out, int out_size, void* d_ws, size_t ws_size,
                              hipStream_t stream) {
  const float* x    = (const float*)d_in[0];
  const float* g1   = (const float*)d_in[1];
  const float* wqkv = (const float*)d_in[2];
  const float* wo   = (const float*)d_in[3];
  const float* g2   = (const float*)d_in[4];
  const float* wr   = (const float*)d_in[5];
  const float* w1   = (const float*)d_in[6];
  const float* w2   = (const float*)d_in[7];
  float* out = (float*)d_out;
  (void)in_sizes; (void)n_in; (void)out_size; (void)ws_size;

  // ---- lifetime-aliased workspace layout (byte offsets) ----
  const size_t MB = (size_t)1 << 20;
  char* ws = (char*)d_ws;
  // region0 [0,40MB): x1(0..8) + qkv(8..32) early;  hb(0..40) late (MoE)
  __bf16* x1    = (__bf16*)(ws + 0 * MB);
  __bf16* qkv   = (__bf16*)(ws + 8 * MB);
  __bf16* hb    = (__bf16*)(ws + 0 * MB);
  // region1 [40,72MB): qh/kh/vh/y2 during attention; w1b (32MB) after
  __bf16* qh    = (__bf16*)(ws + 40 * MB);
  __bf16* khd   = (__bf16*)(ws + 48 * MB);
  __bf16* vhd   = (__bf16*)(ws + 56 * MB);
  __bf16* y2    = (__bf16*)(ws + 64 * MB);
  __bf16* w1b   = (__bf16*)(ws + 40 * MB);
  // region2 [72,80MB): x2
  __bf16* x2    = (__bf16*)(ws + 72 * MB);
  // region3 [80,81MB): router state
  int*    cnt   = (int*)  (ws + 80 * MB);
  int*    tks   = (int*)  (ws + 80 * MB + 4096);
  float*  gts   = (float*)(ws + 80 * MB + 4096 + (size_t)kE * kCAP * 4);
  // region4 [81,101MB): xe
  __bf16* xe    = (__bf16*)(ws + 81 * MB);
  // weights (bf16 copies)
  __bf16* wqkvb = (__bf16*)(ws + 101 * MB);   // 6MB
  __bf16* wob   = (__bf16*)(ws + 107 * MB);   // 2MB
  __bf16* w2b   = (__bf16*)(ws + 109 * MB);   // 32MB  (total 141MB)

  // ---- attention branch ----
  {
    const int n = kD * 3 * kD;
    convert_f32_to_bf16<<<n / 2048, 256, 0, stream>>>(wqkv, wqkvb, n);
  }
  {
    const int n = kD * kD;
    convert_f32_to_bf16<<<n / 2048, 256, 0, stream>>>(wo, wob, n);
  }
  rmsnorm_to_bf16<<<kT, 256, 0, stream>>>(x, g1, x1);
  gemm_bf16_wmma<0><<<dim3(3 * kD / 64, kT / 128, 1), 256, 0, stream>>>(
      x1, wqkvb, qkv, nullptr, nullptr, nullptr, kT, 3 * kD, kD);
  {
    const int tot = kT * kH * (kHD / 2);
    rope_repack<<<(tot + 255) / 256, 256, 0, stream>>>(qkv, qh, khd, vhd);
  }
  attn_wmma<<<kB * kH * (kS / 16), 32, 0, stream>>>(qh, khd, vhd, y2);
  gemm_bf16_wmma<1><<<dim3(kD / 64, kT / 128, 1), 256, 0, stream>>>(
      y2, wob, out, x, nullptr, nullptr, kT, kD, kD);

  // ---- MoE branch ----
  rmsnorm_to_bf16<<<kT, 256, 0, stream>>>(out, g2, x2);
  {  // safe to overwrite q/k/v/y2 region now
    const int n = kE * kD * kFFN;
    convert_f32_to_bf16<<<n / 2048, 256, 0, stream>>>(w1, w1b, n);
    convert_f32_to_bf16<<<n / 2048, 256, 0, stream>>>(w2, w2b, n);
  }
  zero_cnt<<<1, 32, 0, stream>>>(cnt);
  router_topk<<<(kT + 255) / 256, 256, 0, stream>>>(x2, wr, cnt, tks, gts);
  gather_xe<<<kE * kCAP, 256, 0, stream>>>(x2, cnt, tks, gts, xe);
  gemm_bf16_wmma<2><<<dim3(kFFN / 64, kCAP / 128, kE), 256, 0, stream>>>(
      xe, w1b, hb, nullptr, nullptr, nullptr, kCAP, kFFN, kD);
  gemm_bf16_wmma<3><<<dim3(kD / 64, kCAP / 128, kE), 256, 0, stream>>>(
      hb, w2b, out, nullptr, gts, tks, kCAP, kD, kFFN);
}
